// InteractingLayer_11416023073096
// MI455X (gfx1250) — matrix operand order repaired and checked by reading, not compiled, not run on hardware
//
#include <hip/hip_runtime.h>
#include <hip/hip_bf16.h>
#include <math.h>

typedef __attribute__((ext_vector_type(16))) __bf16       v16bf;
typedef __attribute__((ext_vector_type(8)))  float        v8f;
typedef __attribute__((ext_vector_type(4)))  unsigned int u32x4;
typedef __attribute__((ext_vector_type(2)))  unsigned int u32x2;

union ABfrag  { v16bf v; u32x4 q[2]; unsigned short u[16]; };
union Cfrag   { v8f   v; float f[8]; };
union Pack8   { unsigned short u[8];  u32x4 q; };
union Pack16  { unsigned short u[16]; u32x4 q[2]; };
union Pack4   { unsigned short u[4];  u32x2 d; };

__device__ __forceinline__ unsigned short bf16bits(float f) {
    __bf16 h = (__bf16)f;
    unsigned short b;
    __builtin_memcpy(&b, &h, 2);
    return b;
}

__device__ __forceinline__ v8f wmma_bf16(v16bf a, v16bf b, v8f c) {
    // D = A(16x32) * B(32x16) + C, fp32 accumulate
    return __builtin_amdgcn_wmma_f32_16x16x32_bf16(false, a, false, b, (short)0, c, false, false);
}

// ---------------------------------------------------------------------------
// Prep: convert 4 weight matrices (64x128 fp32 row-major) to bf16 in
// fragment-native order: frag f = ((w*2 + k0)*8 + nT),
//   element [lane][j] = W[32*k0 + 16*(lane>>4) + j][16*nT + (lane&15)]
// so the main kernel's B-operand is one contiguous 32-byte chunk per lane.
// ---------------------------------------------------------------------------
__global__ void prep_weights_kernel(const float* __restrict__ Wq,
                                    const float* __restrict__ Wk,
                                    const float* __restrict__ Wv,
                                    const float* __restrict__ Wr,
                                    unsigned short* __restrict__ wsw) {
    int t = blockIdx.x * blockDim.x + threadIdx.x;   // 0 .. 32767
    int j    =  t        & 15;
    int lane = (t >> 4)  & 31;
    int nT   = (t >> 9)  & 7;
    int k0   = (t >> 12) & 1;
    int w    = (t >> 13) & 3;
    const float* W = (w == 0) ? Wq : (w == 1) ? Wk : (w == 2) ? Wv : Wr;
    int row = 32 * k0 + 16 * (lane >> 4) + j;
    int col = 16 * nT + (lane & 15);
    wsw[t] = bf16bits(W[row * 128 + col]);
}

// ---------------------------------------------------------------------------
// Main: one workgroup (256 threads / 8 waves) per batch element.
// ---------------------------------------------------------------------------
__global__ __launch_bounds__(256)
void interacting_layer_kernel(const float* __restrict__ x,
                              const unsigned short* __restrict__ wsw,
                              float* __restrict__ out) {
    // LDS staging (bank-conflict-padded strides); total ~113.6 KB -> 2 WG/WGP
    __shared__ __attribute__((aligned(16))) unsigned short sXP[64 * 72];   // X (phase B) then P (attn)
    __shared__ __attribute__((aligned(16))) unsigned short sQ [64 * 136];  // Q row-major bf16
    __shared__ __attribute__((aligned(16))) unsigned short sK [64 * 136];  // K row-major bf16
    __shared__ __attribute__((aligned(16))) unsigned short sVt[128 * 72];  // V transposed [d][field]
    __shared__ __attribute__((aligned(16))) float          sOUT[64 * 132]; // x@W_R, += attention
    __shared__ __attribute__((aligned(16))) float          sS  [64 * 68];  // per-head scores

    const int tid  = threadIdx.x;
    const int wid  = tid >> 5;
    const int lane = tid & 31;
    const int lhi  = lane >> 4;   // 0/1
    const int llo  = lane & 15;

    const int b = blockIdx.x;
    const float* xb = x + (size_t)b * 64 * 64;

    // ---- Phase A: stage x as bf16 (vectorized: b128 loads, b64 LDS stores) ----
    for (int i = tid * 4; i < 64 * 64; i += 256 * 4) {
        int r = i >> 6, cc = i & 63;
        const float4 v = *(const float4*)(xb + i);
        Pack4 p;
        p.u[0] = bf16bits(v.x); p.u[1] = bf16bits(v.y);
        p.u[2] = bf16bits(v.z); p.u[3] = bf16bits(v.w);
        *(u32x2*)&sXP[r * 72 + cc] = p.d;
    }
    __syncthreads();

    // ---- Phase B: Q,K,V,R projections. nT = wid (runtime); w, mT unrolled so
    //      every store path is branch-free straight-line code. ----
    const int nT = wid;
#pragma unroll
    for (int w = 0; w < 4; ++w) {
#pragma unroll
        for (int mT = 0; mT < 4; ++mT) {
            Cfrag c;
#pragma unroll
            for (int r = 0; r < 8; ++r) c.f[r] = 0.0f;

#pragma unroll
            for (int k0 = 0; k0 < 2; ++k0) {
                ABfrag a, bf;
                int m  = mT * 16 + llo;
                int c0 = k0 * 32 + 16 * lhi;
                a.q[0] = *(const u32x4*)&sXP[m * 72 + c0];
                a.q[1] = *(const u32x4*)&sXP[m * 72 + c0 + 8];
                const unsigned short* wp = wsw + ((((w * 2 + k0) * 8 + nT) * 32 + lane) * 16);
                bf.q[0] = *(const u32x4*)(wp);
                bf.q[1] = *(const u32x4*)(wp + 8);
                c.v = wmma_bf16(a.v, bf.v, c.v);
            }

            if (w <= 1) {                       // Q or K: row-major bf16
                unsigned short* dst = (w == 0) ? sQ : sK;
#pragma unroll
                for (int r = 0; r < 8; ++r) {
                    int row = mT * 16 + r + 8 * lhi;
                    dst[row * 136 + nT * 16 + llo] = bf16bits(c.f[r]);
                }
            } else if (w == 2) {                // V: store transposed, contiguous 16B/lane
                Pack8 p;
#pragma unroll
                for (int r = 0; r < 8; ++r) p.u[r] = bf16bits(c.f[r]);
                *(u32x4*)&sVt[(nT * 16 + llo) * 72 + mT * 16 + 8 * lhi] = p.q;
            } else {                            // R: fp32 into OUT (residual base)
#pragma unroll
                for (int r = 0; r < 8; ++r) {
                    int row = mT * 16 + r + 8 * lhi;
                    sOUT[row * 132 + nT * 16 + llo] = c.f[r];
                }
            }
        }
    }
    __syncthreads();

    // ---- Phase C: attention, one head at a time ----
    for (int h = 0; h < 4; ++h) {
        // scores S = Q_h (64x32) @ K_h^T (32x64): 16 tiles, 2 per wave
#pragma unroll
        for (int it = 0; it < 2; ++it) {
            int tile = wid * 2 + it;
            int mT = tile >> 2, nTs = tile & 3;
            ABfrag a, bb;
            Cfrag c;
#pragma unroll
            for (int r = 0; r < 8; ++r) c.f[r] = 0.0f;

            int m  = mT * 16 + llo;
            int c0 = h * 32 + 16 * lhi;
            a.q[0] = *(const u32x4*)&sQ[m * 136 + c0];
            a.q[1] = *(const u32x4*)&sQ[m * 136 + c0 + 8];

            int jrow = nTs * 16 + llo;           // B column = K row (k[j,d])
            int kc   = h * 32 + 16 * lhi;
            bb.q[0] = *(const u32x4*)&sK[jrow * 136 + kc];
            bb.q[1] = *(const u32x4*)&sK[jrow * 136 + kc + 8];

            c.v = wmma_bf16(a.v, bb.v, c.v);
#pragma unroll
            for (int r = 0; r < 8; ++r)
                sS[(mT * 16 + r + 8 * lhi) * 68 + nTs * 16 + llo] = c.f[r];
        }
        __syncthreads();

        // softmax: 4 lanes per row (16 elements each), shuffle-reduce max/sum,
        // write bf16 P into sXP (X is dead). All 256 threads active.
        {
            int row = tid >> 2, seg = tid & 3;
            float* srow = &sS[row * 68 + seg * 16];
            float mx = -INFINITY;
#pragma unroll
            for (int j = 0; j < 16; ++j) mx = fmaxf(mx, srow[j]);
            mx = fmaxf(mx, __shfl_xor(mx, 1));
            mx = fmaxf(mx, __shfl_xor(mx, 2));
            float e[16], sum = 0.0f;
#pragma unroll
            for (int j = 0; j < 16; ++j) { e[j] = __expf(srow[j] - mx); sum += e[j]; }
            sum += __shfl_xor(sum, 1);
            sum += __shfl_xor(sum, 2);
            float inv = 1.0f / sum;
            Pack16 p;
#pragma unroll
            for (int j = 0; j < 16; ++j) p.u[j] = bf16bits(e[j] * inv);
            u32x4* dst = (u32x4*)&sXP[row * 72 + seg * 16];
            dst[0] = p.q[0];
            dst[1] = p.q[1];
        }
        __syncthreads();

        // out_h = P (64x64) @ V_h (64x32), accumulate into OUT: 8 tiles, 1/wave
        {
            int mT = wid >> 1, nTo = wid & 1;
            Cfrag c;
#pragma unroll
            for (int r = 0; r < 8; ++r)
                c.f[r] = sOUT[(mT * 16 + r + 8 * lhi) * 132 + h * 32 + nTo * 16 + llo];
#pragma unroll
            for (int k0 = 0; k0 < 2; ++k0) {
                ABfrag a, bb;
                int m  = mT * 16 + llo;
                int c0 = k0 * 32 + 16 * lhi;
                a.q[0] = *(const u32x4*)&sXP[m * 72 + c0];
                a.q[1] = *(const u32x4*)&sXP[m * 72 + c0 + 8];
                int d  = h * 32 + nTo * 16 + llo;    // B column = output head-dim
                int kc = k0 * 32 + 16 * lhi;
                bb.q[0] = *(const u32x4*)&sVt[d * 72 + kc];
                bb.q[1] = *(const u32x4*)&sVt[d * 72 + kc + 8];
                c.v = wmma_bf16(a.v, bb.v, c.v);
            }
#pragma unroll
            for (int r = 0; r < 8; ++r)
                sOUT[(mT * 16 + r + 8 * lhi) * 132 + h * 32 + nTo * 16 + llo] = c.f[r];
        }
        __syncthreads();
    }

    // ---- Phase D: ReLU + vectorized coalesced store (b128) ----
    float* ob = out + (size_t)b * 64 * 128;
    for (int i = tid * 4; i < 64 * 128; i += 256 * 4) {
        int r = i >> 7, cc = i & 127;
        float4 v = *(const float4*)&sOUT[r * 132 + cc];
        v.x = fmaxf(v.x, 0.0f);
        v.y = fmaxf(v.y, 0.0f);
        v.z = fmaxf(v.z, 0.0f);
        v.w = fmaxf(v.w, 0.0f);
        *(float4*)(ob + i) = v;
    }
}

extern "C" void kernel_launch(void* const* d_in, const int* in_sizes, int n_in,
                              void* d_out, int out_size, void* d_ws, size_t ws_size,
                              hipStream_t stream) {
    (void)in_sizes; (void)n_in; (void)out_size; (void)ws_size;
    const float* x  = (const float*)d_in[0];
    const float* Wq = (const float*)d_in[1];
    const float* Wk = (const float*)d_in[2];
    const float* Wv = (const float*)d_in[3];
    const float* Wr = (const float*)d_in[4];
    float* out = (float*)d_out;
    unsigned short* wsw = (unsigned short*)d_ws;   // 32768 bf16 halves = 64 KB

    prep_weights_kernel<<<128, 256, 0, stream>>>(Wq, Wk, Wv, Wr, wsw);
    interacting_layer_kernel<<<8192, 256, 0, stream>>>(x, wsw, out);
}
